// BiAttention_45818711114387
// MI455X (gfx1250) — matrix operand order repaired and checked by reading, not compiled, not run on hardware
//
#include <hip/hip_runtime.h>
#include <hip/hip_bf16.h>

typedef __attribute__((ext_vector_type(16))) __bf16 v16bf;
typedef __attribute__((ext_vector_type(8)))  __bf16 v8bf;
typedef __attribute__((ext_vector_type(8)))  float  v8f;

#define B_  8
#define CL  4096
#define QL  512
#define D_  256

static __device__ __forceinline__ v16bf mk16(v8bf lo, v8bf hi) {
  v16bf r;
#pragma unroll
  for (int i = 0; i < 8; i++) { r[i] = lo[i]; r[i + 8] = hi[i]; }
  return r;
}
// monotonic unsigned key for float (for LDS atomic max on floats)
static __device__ __forceinline__ unsigned fkey(float f) {
  unsigned u = __float_as_uint(f);
  return (u & 0x80000000u) ? ~u : (u | 0x80000000u);
}
static __device__ __forceinline__ float funkey(unsigned k) {
  return (k & 0x80000000u) ? __uint_as_float(k & 0x7fffffffu) : __uint_as_float(~k);
}
static __device__ __forceinline__ v8f wmma_bf16(v16bf a, v16bf b, v8f c) {
  return __builtin_amdgcn_wmma_f32_16x16x32_bf16(false, a, false, b, (short)0, c,
                                                 false, false);
}

// ---------------- prep: q -> bf16 (row major + transposed), qw = q . w_q -----
__global__ __launch_bounds__(128)
void bi_prep(const float* __restrict__ q, const float* __restrict__ w,
             __bf16* __restrict__ qb, __bf16* __restrict__ qt,
             float* __restrict__ qw) {
  __shared__ float qwL[16];
  const int blk = blockIdx.x;            // 256 blocks: 16 q-rows each
  const int b = blk >> 5, j0 = (blk & 31) * 16;
  const int t = threadIdx.x, r = t >> 3, d0 = (t & 7) * 32;
  const int j = j0 + r;
  if (t < 16) qwL[t] = 0.f;
  __syncthreads();
  const float* qrow = q + ((size_t)b * QL + j) * D_;
  float part = 0.f;
#pragma unroll 8
  for (int k = 0; k < 32; k++) {
    float v = qrow[d0 + k];
    part += v * w[d0 + k];
    __bf16 bv = (__bf16)v;
    qb[((size_t)b * QL + j) * D_ + d0 + k] = bv;
    qt[((size_t)b * D_ + d0 + k) * QL + j] = bv;
  }
  atomicAdd(&qwL[r], part);
  __syncthreads();
  if ((t & 7) == 0) qw[b * QL + j] = qwL[r];
}

// ---------------- fused: S strip -> softmax -> P@Q, writes out[:, 0:768] -----
__global__ __launch_bounds__(128)
void bi_main(const float* __restrict__ c, const float* __restrict__ w,
             const __bf16* __restrict__ qb, const __bf16* __restrict__ qt,
             const float* __restrict__ qw, float* __restrict__ out,
             float* __restrict__ rowmaxG) {
  __shared__ __align__(16) __bf16 cmL[16][D_];   // (c * w_m) strip, bf16
  __shared__ float  sL[16][QL];                  // S strip (then exp values)
  __shared__ __align__(16) __bf16 pL[16][QL];    // normalized P, bf16
  __shared__ float  cwL[16];
  __shared__ float  qwL[QL];
  __shared__ unsigned rmK[16];
  __shared__ float  rsum[16];

  const int b = blockIdx.y;
  const int i0 = blockIdx.x * 16;
  const int t = threadIdx.x;
  const int wv = t >> 5;
  const int lane = t & 31;
  const int lrow = lane & 15;
  const int lgrp = lane >> 4;

  // ---- phase 0: stage c*w_m (bf16) + cw dot + qw row into LDS
  if (t < 16) { cwL[t] = 0.f; rmK[t] = 0u; rsum[t] = 0.f; }
#pragma unroll
  for (int k = 0; k < 4; k++) qwL[t * 4 + k] = qw[b * QL + t * 4 + k];
  __syncthreads();
  {
    const int r = t >> 3, d0 = (t & 7) * 32;
    const float* crow = c + ((size_t)b * CL + i0 + r) * D_;
    float part = 0.f;
#pragma unroll 8
    for (int k = 0; k < 32; k++) {
      float cv = crow[d0 + k];
      cmL[r][d0 + k] = (__bf16)(cv * w[2 * D_ + d0 + k]);
      part += cv * w[D_ + d0 + k];
    }
    atomicAdd(&cwL[r], part);
  }
  __syncthreads();

  // ---- phase 1: S(16x512) = Cm(16x256) x Q^T via WMMA bf16
  v16bf afrag[8];
#pragma unroll
  for (int kc = 0; kc < 8; kc++) {
    int kb = kc * 32 + lgrp * 8;                 // A layout: K {kb..kb+7, kb+16..kb+23}
    afrag[kc] = mk16(*(const v8bf*)&cmL[lrow][kb], *(const v8bf*)&cmL[lrow][kb + 16]);
  }
  for (int jt = wv; jt < 32; jt += 4) {          // 8 j-tiles per wave
    v8f acc = {};
    const __bf16* qrow = qb + ((size_t)b * QL + jt * 16 + lrow) * D_;
#pragma unroll
    for (int kc = 0; kc < 8; kc++) {
      int kb2 = kc * 32 + lgrp * 8;              // B layout: K {kb2..kb2+7, kb2+16..kb2+23}
      v16bf bfrag = mk16(*(const v8bf*)(qrow + kb2), *(const v8bf*)(qrow + kb2 + 16));
      acc = wmma_bf16(afrag[kc], bfrag, acc);
    }
    float qwv = qwL[jt * 16 + lrow];
#pragma unroll
    for (int r = 0; r < 8; r++) {
      int mr = lgrp * 8 + r;
      sL[mr][jt * 16 + lrow] = acc[r] + cwL[mr] + qwv;
    }
  }
  __syncthreads();

  // ---- softmax over j (each thread: one row, 64 columns)
  const int rr = t >> 3, cc = (t & 7) * 64;
  float lm = -3.0e38f;
#pragma unroll 8
  for (int k = 0; k < 64; k++) lm = fmaxf(lm, sL[rr][cc + k]);
  atomicMax(&rmK[rr], fkey(lm));
  __syncthreads();
  const float m = funkey(rmK[rr]);
  float ls = 0.f;
#pragma unroll 8
  for (int k = 0; k < 64; k++) {
    float e = __expf(sL[rr][cc + k] - m);
    sL[rr][cc + k] = e;
    ls += e;
  }
  atomicAdd(&rsum[rr], ls);
  __syncthreads();
  const float inv = 1.f / rsum[rr];
#pragma unroll 8
  for (int k = 0; k < 64; k++) pL[rr][cc + k] = (__bf16)(sL[rr][cc + k] * inv);
  if ((t & 7) == 0) rowmaxG[b * CL + i0 + rr] = m;   // exact row-max for b_attn
  __syncthreads();

  // ---- phase 2: c2q(16x256) = P(16x512) x Q(512x256) via WMMA bf16
  v8f zero = {};
  v8f acc2[4] = {zero, zero, zero, zero};
  for (int kc = 0; kc < 16; kc++) {
    int kb = kc * 32 + lgrp * 8;
    v16bf pa = mk16(*(const v8bf*)&pL[lrow][kb], *(const v8bf*)&pL[lrow][kb + 16]);
    int kb2 = kc * 32 + lgrp * 8;
#pragma unroll
    for (int dtl = 0; dtl < 4; dtl++) {
      int d = (wv * 4 + dtl) * 16 + lrow;
      const __bf16* qtrow = qt + ((size_t)b * D_ + d) * QL;
      v16bf bfrag = mk16(*(const v8bf*)(qtrow + kb2), *(const v8bf*)(qtrow + kb2 + 16));
      acc2[dtl] = wmma_bf16(pa, bfrag, acc2[dtl]);
    }
  }
  // ---- epilogue: quarters 1..3 of the output
#pragma unroll
  for (int dtl = 0; dtl < 4; dtl++) {
    int dcol = (wv * 4 + dtl) * 16 + lrow;
#pragma unroll
    for (int r = 0; r < 8; r++) {
      int mr = lgrp * 8 + r;
      size_t gi = (size_t)b * CL + i0 + mr;
      float c2q = acc2[dtl][r];
      float cv = c[gi * D_ + dcol];
      float* orow = out + gi * (4 * D_);
      orow[dcol] = cv;
      orow[D_ + dcol] = c2q;
      orow[2 * D_ + dcol] = cv * c2q;
    }
  }
}

// ---------------- b_attn = softmax_i(rowmax); q2c = b_attn . c ---------------
__global__ __launch_bounds__(256)
void bi_battn(const float* __restrict__ rowmaxG, const float* __restrict__ c,
              float* __restrict__ q2c) {
  __shared__ float e[CL];
  __shared__ unsigned mk;
  __shared__ float ssum;
  const int b = blockIdx.x, t = threadIdx.x;
  if (t == 0) { mk = 0u; ssum = 0.f; }
  __syncthreads();
  float lm = -3.0e38f;
  for (int i = t; i < CL; i += 256) lm = fmaxf(lm, rowmaxG[b * CL + i]);
  atomicMax(&mk, fkey(lm));
  __syncthreads();
  const float m = funkey(mk);
  float ls = 0.f;
  for (int i = t; i < CL; i += 256) {
    float ev = __expf(rowmaxG[b * CL + i] - m);
    e[i] = ev;
    ls += ev;
  }
  atomicAdd(&ssum, ls);
  __syncthreads();
  const float inv = 1.f / ssum;
  float acc = 0.f;
  const float* cb = c + (size_t)b * CL * D_ + t;    // thread t owns column d=t
  for (int i = 0; i < CL; i++) acc += e[i] * cb[(size_t)i * D_];
  q2c[b * D_ + t] = acc * inv;
}

// ---------------- out[:, 768:1024] = c * q2c (streaming, float4) -------------
__global__ __launch_bounds__(256)
void bi_q4(const float* __restrict__ c, const float* __restrict__ q2c,
           float* __restrict__ out) {
  size_t idx = (size_t)blockIdx.x * 256 + threadIdx.x;  // B*CL*64 float4 chunks
  if (idx >= (size_t)B_ * CL * 64) return;
  size_t rowi = idx >> 6;
  int dv = (int)(idx & 63) * 4;
  int b = (int)(rowi >> 12);
  const float4 cv = *(const float4*)(c + rowi * D_ + dv);
  const float4 qv = *(const float4*)(q2c + b * D_ + dv);
  float4 o;
  o.x = cv.x * qv.x; o.y = cv.y * qv.y; o.z = cv.z * qv.z; o.w = cv.w * qv.w;
  *(float4*)(out + rowi * (4 * D_) + 3 * D_ + dv) = o;
}

extern "C" void kernel_launch(void* const* d_in, const int* in_sizes, int n_in,
                              void* d_out, int out_size, void* d_ws, size_t ws_size,
                              hipStream_t stream) {
  const float* q = (const float*)d_in[0];
  const float* c = (const float*)d_in[1];
  const float* w = (const float*)d_in[2];
  float* out = (float*)d_out;
  char* ws = (char*)d_ws;
  // workspace layout (≈ 4.15 MB total)
  __bf16* qb  = (__bf16*)(ws);                               // 2 MB
  __bf16* qt  = (__bf16*)(ws + 2097152);                     // 2 MB
  float*  qw  = (float*)(ws + 4194304);                      // 16 KB
  float*  rm  = (float*)(ws + 4194304 + 16384);              // 128 KB
  float*  q2c = (float*)(ws + 4194304 + 16384 + 131072);     // 8 KB

  bi_prep <<<dim3(256),         dim3(128), 0, stream>>>(q, w, qb, qt, qw);
  bi_main <<<dim3(CL / 16, B_), dim3(128), 0, stream>>>(c, w, qb, qt, qw, out, rm);
  bi_battn<<<dim3(B_),          dim3(256), 0, stream>>>(rm, c, q2c);
  bi_q4   <<<dim3(8192),        dim3(256), 0, stream>>>(c, q2c, out);
}
